// SelfAttention_29824252903917
// MI455X (gfx1250) — matrix operand order repaired and checked by reading, not compiled, not run on hardware
//
#include <hip/hip_runtime.h>
#include <hip/hip_bf16.h>

// ---------------------------------------------------------------------------
// Problem constants (from reference): B=8, H=W=56, C=384, HEADS=6, DIM=384
// Downsampled tokens L = T = 28*28 = 784, dh = 64.
// ---------------------------------------------------------------------------
#define BATCH   8
#define CH      384
#define HIN     56
#define WIN     56
#define HQ      28
#define WQ      28
#define LTOK    784      // 28*28
#define TTILES  49       // 784 / 16
#define HEADS   6
#define DH      64
#define DIM     384
#define SCALE_QK 0.05103103630798287f   // 384^-0.5, folded into q-GEMM

typedef __attribute__((ext_vector_type(16))) _Float16 v16h;
typedef __attribute__((ext_vector_type(8)))  float    v8f;

// ---------------------------------------------------------------------------
// WMMA fragment loaders (16-bit, 16x16x32); A/B use this file's own contiguous k order.
// A (16x32, MxK): lane group g=lane/16, row M = lane&15.
//   VGPR v in 0..7 -> K = g*16 + 2v + h (contiguous, matches B)
// B (32x16, KxN): lane group g, col N = lane&15, K = g*16 + 2v + h (contiguous).
// C/D (16x16 f32): VGPR r -> M = r + 8*g, N = lane&15.
// ---------------------------------------------------------------------------
__device__ __forceinline__ void load_a_f16(v16h& af, const _Float16* __restrict__ A,
                                           int lda, int m0, int k0, int lane) {
  int g = lane >> 4;
  const _Float16* base = A + (long)(m0 + (lane & 15)) * lda + k0 + g * 16;
#pragma unroll
  for (int v = 0; v < 4; ++v) { af[2*v] = base[2*v]; af[2*v+1] = base[2*v+1]; }
  const _Float16* base2 = base + 8;
#pragma unroll
  for (int v = 0; v < 4; ++v) { af[8+2*v] = base2[2*v]; af[9+2*v] = base2[2*v+1]; }
}

__device__ __forceinline__ void load_b_f16(v16h& bf, const _Float16* __restrict__ Bm,
                                           int ldb, int n0, int k0, int lane) {
  int g = lane >> 4;
  const _Float16* base = Bm + (long)(n0 + (lane & 15)) * ldb + k0 + g * 16;
#pragma unroll
  for (int j = 0; j < 16; ++j) bf[j] = base[j];
}

// B from an f32 weight matrix W (N x K row-major), computing A @ W^T: B[k][n]=W[n][k]
__device__ __forceinline__ void load_b_f32w(v16h& bf, const float* __restrict__ W,
                                            int ldw, int n0, int k0, int lane) {
  int g = lane >> 4;
  const float* base = W + (long)(n0 + (lane & 15)) * ldw + k0 + g * 16;
#pragma unroll
  for (int j = 0; j < 16; ++j) bf[j] = (_Float16)base[j];
}

// Bounds-checked B loader (zero-fill K tail) for the T=784 (not %32) K-loop.
__device__ __forceinline__ void load_b_f16_bc(v16h& bf, const _Float16* __restrict__ Bm,
                                              int ldb, int n0, int k0, int kmax, int lane) {
  int g = lane >> 4;
  const _Float16* row = Bm + (long)(n0 + (lane & 15)) * ldb;
#pragma unroll
  for (int j = 0; j < 16; ++j) {
    int v = j >> 1, h = j & 1;
    int kk = k0 + g*16 + 2*v + h;
    bf[j] = (kk < kmax) ? row[kk] : (_Float16)0.0f;
  }
}

// A fragment built from an LDS row of f32 probabilities (convert to f16 in flight).
__device__ __forceinline__ void load_a_lds_bc(v16h& af, const float* row,
                                              int k0, int kmax, int lane) {
  int g = lane >> 4;
#pragma unroll
  for (int j = 0; j < 16; ++j) {
    int v = j >> 1, h = j & 1;
    int k = g*16 + 2*v + h;
    int kk = k0 + k;
    af[j] = (kk < kmax) ? (_Float16)row[kk] : (_Float16)0.0f;
  }
}

__device__ __forceinline__ v8f wmma_f16(const v16h& a, const v16h& b, const v8f& c) {
  return __builtin_amdgcn_wmma_f32_16x16x32_f16(false, a, false, b, (short)0, c, false, false);
}

// ---------------------------------------------------------------------------
// Kernel 1: depthwise 3x3 stride-2 conv + LayerNorm -> q_ln (f16),
//           2x2 avg pool -> kv (f16).  One block per (b, token), 128 threads.
// ---------------------------------------------------------------------------
__global__ void preprocess_kernel(const float* __restrict__ x, const float* __restrict__ convw,
                                  const float* __restrict__ ln_g, const float* __restrict__ ln_b,
                                  _Float16* __restrict__ q_ln, _Float16* __restrict__ kv) {
  int blk = blockIdx.x;
  int b = blk / LTOK, l = blk % LTOK;
  int oy = l / WQ, ox = l % WQ;
  const float* xb = x + (long)b * (HIN * WIN) * CH;
  __shared__ float red[128];
  __shared__ float stats[2];
  int tid = threadIdx.x;

  float qv[3], kvv[3];
  float psum = 0.f, psum2 = 0.f;
#pragma unroll
  for (int i = 0; i < 3; ++i) {
    int c = tid + i * 128;
    float acc = 0.f;
#pragma unroll
    for (int ky = 0; ky < 3; ++ky) {
      int iy = 2 * oy - 1 + ky;
      if (iy < 0 || iy >= HIN) continue;
#pragma unroll
      for (int kx = 0; kx < 3; ++kx) {
        int ix = 2 * ox - 1 + kx;
        if (ix < 0 || ix >= WIN) continue;
        acc += convw[c * 9 + ky * 3 + kx] * xb[(long)(iy * WIN + ix) * CH + c];
      }
    }
    qv[i] = acc; psum += acc; psum2 += acc * acc;
    int py = 2 * oy, px = 2 * ox;
    kvv[i] = 0.25f * (xb[(long)(py * WIN + px) * CH + c] +
                      xb[(long)(py * WIN + px + 1) * CH + c] +
                      xb[(long)((py + 1) * WIN + px) * CH + c] +
                      xb[(long)((py + 1) * WIN + px + 1) * CH + c]);
  }
  red[tid] = psum; __syncthreads();
  for (int s = 64; s > 0; s >>= 1) { if (tid < s) red[tid] += red[tid + s]; __syncthreads(); }
  if (tid == 0) stats[0] = red[0] / (float)CH;
  __syncthreads();
  red[tid] = psum2; __syncthreads();
  for (int s = 64; s > 0; s >>= 1) { if (tid < s) red[tid] += red[tid + s]; __syncthreads(); }
  if (tid == 0) {
    float mu = stats[0];
    stats[1] = rsqrtf(red[0] / (float)CH - mu * mu + 1e-5f);
  }
  __syncthreads();
  float mu = stats[0], rstd = stats[1];
  _Float16* qo = q_ln + ((long)b * LTOK + l) * CH;
  _Float16* ko = kv   + ((long)b * LTOK + l) * CH;
#pragma unroll
  for (int i = 0; i < 3; ++i) {
    int c = tid + i * 128;
    qo[c] = (_Float16)(((qv[i] - mu) * rstd) * ln_g[c] + ln_b[c]);
    ko[c] = (_Float16)kvv[i];
  }
}

// ---------------------------------------------------------------------------
// Kernel 2: out = A(f16, M x K) @ W(f32, N x K)^T, f16 output, optional
// transposed store (for V^T) and scale epilogue. One wave per 16x16 tile.
// ---------------------------------------------------------------------------
__global__ void gemm_awT_f16(const _Float16* __restrict__ A, const float* __restrict__ W,
                             _Float16* __restrict__ Out, int M, int N, int K,
                             long aStride, long oStride, float scale, int transposeStore) {
  int lane = threadIdx.x & 31;
  int n0 = blockIdx.x * 16, m0 = blockIdx.y * 16, b = blockIdx.z;
  const _Float16* Ab = A + (long)b * aStride;
  v8f acc = {};
  for (int k0 = 0; k0 < K; k0 += 32) {
    v16h af, bf;
    load_a_f16(af, Ab, K, m0, k0, lane);
    load_b_f32w(bf, W, K, n0, k0, lane);
    acc = wmma_f16(af, bf, acc);
  }
  _Float16* Ob = Out + (long)b * oStride;
  int g = lane >> 4, nn = lane & 15;
  if (!transposeStore) {
#pragma unroll
    for (int r = 0; r < 8; ++r)
      Ob[(long)(m0 + r + 8 * g) * N + n0 + nn] = (_Float16)(acc[r] * scale);
  } else {
#pragma unroll
    for (int r = 0; r < 8; ++r)
      Ob[(long)(n0 + nn) * M + (m0 + r + 8 * g)] = (_Float16)(acc[r] * scale);
  }
}

// ---------------------------------------------------------------------------
// Kernel 4: projection GEMM, f32 output + bias -> d_out.
// ---------------------------------------------------------------------------
__global__ void gemm_awT_f32_bias(const _Float16* __restrict__ A, const float* __restrict__ W,
                                  const float* __restrict__ bias, float* __restrict__ Out,
                                  int M, int N, int K, long aStride, long oStride) {
  int lane = threadIdx.x & 31;
  int n0 = blockIdx.x * 16, m0 = blockIdx.y * 16, b = blockIdx.z;
  const _Float16* Ab = A + (long)b * aStride;
  v8f acc = {};
  for (int k0 = 0; k0 < K; k0 += 32) {
    v16h af, bf;
    load_a_f16(af, Ab, K, m0, k0, lane);
    load_b_f32w(bf, W, K, n0, k0, lane);
    acc = wmma_f16(af, bf, acc);
  }
  float* Ob = Out + (long)b * oStride;
  int g = lane >> 4, nn = lane & 15;
  float bv = bias[n0 + nn];
#pragma unroll
  for (int r = 0; r < 8; ++r)
    Ob[(long)(m0 + r + 8 * g) * N + n0 + nn] = acc[r] + bv;
}

// ---------------------------------------------------------------------------
// Kernel 3 (fused attention core): one block per (b, 16-row l-block).
// Exploits MI455X's 320 KB LDS: the full 6-head x 16-row x 784-key score slab
// (301 KB f32) lives in LDS end-to-end:
//   A) QK^T WMMA for all heads/t-tiles -> LDS
//   B) bilinear-upsampled res_attn + cross-head w_fuse mix, in place
//   C) row softmax (wave shuffle reductions), normalized probs in place
//   D) P@V WMMA, A-fragments converted f32->f16 straight out of LDS
// No global score/prob intermediates at all.
// ---------------------------------------------------------------------------
__device__ __forceinline__ void up_coord(int p, int& i0, int& i1, float& w) {
  float c = 0.5f * (float)p - 0.25f;
  c = fminf(fmaxf(c, 0.0f), 13.0f);
  float f = floorf(c);
  i0 = (int)f; i1 = min(i0 + 1, 13);
  w = c - f;
}

__global__ void __launch_bounds__(256)
fused_attn_kernel(const _Float16* __restrict__ qh, const _Float16* __restrict__ kh,
                  const _Float16* __restrict__ vT, const float* __restrict__ res_attn,
                  const float* __restrict__ w_fuse, const float* __restrict__ b_fuse,
                  _Float16* __restrict__ ctx) {
  extern __shared__ float S[];          // [HEADS][16][LTOK]  = 301,056 bytes
  __shared__ float sWF[HEADS * 12];     // fuse weights
  __shared__ float sBF[HEADS];

  int l0 = blockIdx.x * 16;
  int b  = blockIdx.y;
  int tid = threadIdx.x, wave = tid >> 5, lane = tid & 31;

  if (tid < HEADS * 12) sWF[tid] = w_fuse[tid];
  if (tid < HEADS)      sBF[tid] = b_fuse[tid];

  const _Float16* Qb = qh + (long)b * LTOK * DIM;
  const _Float16* Kb = kh + (long)b * LTOK * DIM;

  // ---- Stage A: raw QK^T scores for all heads ---------------------------
  for (int idx = wave; idx < HEADS * TTILES; idx += 8) {
    int h = idx / TTILES, t0 = (idx % TTILES) * 16;
    __builtin_prefetch(Kb + (long)(t0 + 16) * DIM + h * DH, 0, 1);  // global_prefetch_b8
    v8f acc = {};
#pragma unroll
    for (int kb = 0; kb < DH; kb += 32) {
      v16h af, bf;
      load_a_f16(af, Qb, DIM, l0, h * DH + kb, lane);
      load_b_f16(bf, Kb, DIM, t0, h * DH + kb, lane);
      acc = wmma_f16(af, bf, acc);
    }
    int g = lane >> 4, nn = lane & 15;
    float* Sh = S + h * 16 * LTOK;
#pragma unroll
    for (int r = 0; r < 8; ++r) Sh[(r + 8 * g) * LTOK + t0 + nn] = acc[r];
  }
  __syncthreads();

  // ---- Stage B: bilinear residual + cross-head fuse (in place) ----------
  for (int p = tid; p < 16 * LTOK; p += 256) {
    int li = p / LTOK, t = p % LTOK;
    int l = l0 + li;
    int y = l / WQ, xq = l % WQ;
    int y0, y1, x0, x1; float wy, wx;
    up_coord(y, y0, y1, wy);
    up_coord(xq, x0, x1, wx);

    float sv[HEADS], rx[HEADS];
#pragma unroll
    for (int h = 0; h < HEADS; ++h) sv[h] = S[(h * 16 + li) * LTOK + t];
#pragma unroll
    for (int h = 0; h < HEADS; ++h) {
      const float* R = res_attn + ((long)(b * HEADS + h) * 196) * LTOK + t;
      float r00 = R[(long)(y0 * 14 + x0) * LTOK];
      float r01 = R[(long)(y0 * 14 + x1) * LTOK];
      float r10 = R[(long)(y1 * 14 + x0) * LTOK];
      float r11 = R[(long)(y1 * 14 + x1) * LTOK];
      rx[h] = (1.f - wy) * ((1.f - wx) * r00 + wx * r01) +
              wy * ((1.f - wx) * r10 + wx * r11);
    }
#pragma unroll
    for (int o = 0; o < HEADS; ++o) {
      float acc = sBF[o];
#pragma unroll
      for (int h = 0; h < HEADS; ++h)
        acc += sWF[o * 12 + h] * sv[h] + sWF[o * 12 + HEADS + h] * rx[h];
      S[(o * 16 + li) * LTOK + t] = acc;
    }
  }
  __syncthreads();

  // ---- Stage C: softmax per (head,row), normalized probs in place -------
  for (int r = wave; r < HEADS * 16; r += 8) {
    float* row = S + r * LTOK;
    float m = -3.4e38f;
    for (int t = lane; t < LTOK; t += 32) m = fmaxf(m, row[t]);
#pragma unroll
    for (int off = 16; off > 0; off >>= 1) m = fmaxf(m, __shfl_xor(m, off, 32));
    float ssum = 0.f;
    for (int t = lane; t < LTOK; t += 32) { float e = __expf(row[t] - m); row[t] = e; ssum += e; }
#pragma unroll
    for (int off = 16; off > 0; off >>= 1) ssum += __shfl_xor(ssum, off, 32);
    float inv = 1.0f / ssum;
    for (int t = lane; t < LTOK; t += 32) row[t] *= inv;
  }
  __syncthreads();

  // ---- Stage D: P @ V, A-fragments straight from LDS --------------------
  const _Float16* Vb = vT + (long)b * DIM * LTOK;
  _Float16* Cb = ctx + (long)b * LTOK * DIM;
  for (int idx = wave; idx < HEADS * (DH / 16); idx += 8) {
    int o = idx >> 2, n0 = (idx & 3) * 16;
    const float* Prow = S + (o * 16 + (lane & 15)) * LTOK;
    const _Float16* Vh = Vb + (long)(o * DH) * LTOK;
    v8f acc = {};
    for (int k0 = 0; k0 < LTOK; k0 += 32) {
      v16h af, bf;
      load_a_lds_bc(af, Prow, k0, LTOK, lane);
      if (k0 + 32 <= LTOK) {
        __builtin_prefetch(Vh + (long)(n0 + (lane & 15)) * LTOK + k0 + 32, 0, 1);
        load_b_f16(bf, Vh, LTOK, n0, k0, lane);
      } else {
        load_b_f16_bc(bf, Vh, LTOK, n0, k0, LTOK, lane);
      }
      acc = wmma_f16(af, bf, acc);
    }
    int g = lane >> 4, nn = lane & 15;
#pragma unroll
    for (int r = 0; r < 8; ++r)
      Cb[(long)(l0 + r + 8 * g) * DIM + o * DH + n0 + nn] = (_Float16)acc[r];
  }
}

// ---------------------------------------------------------------------------
// Host launcher
// ---------------------------------------------------------------------------
extern "C" void kernel_launch(void* const* d_in, const int* in_sizes, int n_in,
                              void* d_out, int out_size, void* d_ws, size_t ws_size,
                              hipStream_t stream) {
  (void)in_sizes; (void)n_in; (void)out_size; (void)ws_size;
  const float* x        = (const float*)d_in[0];
  const float* res_attn = (const float*)d_in[1];
  const float* conv_q_w = (const float*)d_in[2];
  const float* ln_g     = (const float*)d_in[3];
  const float* ln_b     = (const float*)d_in[4];
  const float* wq       = (const float*)d_in[5];
  const float* wk       = (const float*)d_in[6];
  const float* wv       = (const float*)d_in[7];
  const float* w_proj   = (const float*)d_in[8];
  const float* b_proj   = (const float*)d_in[9];
  const float* w_fuse   = (const float*)d_in[10];
  const float* b_fuse   = (const float*)d_in[11];
  float* out = (float*)d_out;

  // Workspace partition (~29 MB total; no global score/prob intermediates)
  char* ws = (char*)d_ws;
  size_t off = 0;
  auto carve = [&](size_t bytes) -> void* {
    void* p = ws + off;
    off += (bytes + 255) & ~(size_t)255;
    return p;
  };
  const long BLD = (long)BATCH * LTOK * DIM;            // 2,408,448 elements
  _Float16* q_ln = (_Float16*)carve(BLD * sizeof(_Float16));
  _Float16* kv   = (_Float16*)carve(BLD * sizeof(_Float16));
  _Float16* qh   = (_Float16*)carve(BLD * sizeof(_Float16));
  _Float16* kh   = (_Float16*)carve(BLD * sizeof(_Float16));
  _Float16* vT   = (_Float16*)carve(BLD * sizeof(_Float16));   // (B, DIM, LTOK)
  _Float16* ctx  = (_Float16*)carve(BLD * sizeof(_Float16));

  // 1) conv+LN and avgpool
  preprocess_kernel<<<BATCH * LTOK, 128, 0, stream>>>(x, conv_q_w, ln_g, ln_b, q_ln, kv);

  // 2) QKV projections (scale folded into q; V stored transposed)
  dim3 ggrid(DIM / 16, LTOK / 16, BATCH);
  gemm_awT_f16<<<ggrid, 32, 0, stream>>>(q_ln, wq, qh, LTOK, DIM, DIM,
                                         (long)LTOK * DIM, (long)LTOK * DIM, SCALE_QK, 0);
  gemm_awT_f16<<<ggrid, 32, 0, stream>>>(kv, wk, kh, LTOK, DIM, DIM,
                                         (long)LTOK * DIM, (long)LTOK * DIM, 1.0f, 0);
  gemm_awT_f16<<<ggrid, 32, 0, stream>>>(kv, wv, vT, LTOK, DIM, DIM,
                                         (long)LTOK * DIM, (long)DIM * LTOK, 1.0f, 1);

  // 3) fused scores + residual-fuse + softmax + P@V, score slab in 301KB LDS
  const size_t SMEM = (size_t)HEADS * 16 * LTOK * sizeof(float);  // 301,056 B
  dim3 agrid(LTOK / 16, BATCH);
  fused_attn_kernel<<<agrid, 256, SMEM, stream>>>(qh, kh, vT, res_attn, w_fuse, b_fuse, ctx);

  // 4) output projection + bias -> f32 d_out
  gemm_awT_f32_bias<<<ggrid, 32, 0, stream>>>(ctx, w_proj, b_proj, out, LTOK, DIM, DIM,
                                              (long)LTOK * DIM, (long)LTOK * DIM);
}
